// Mimo_VisionTransformer_15642270892550
// MI455X (gfx1250) — matrix operand order repaired and checked by reading, not compiled, not run on hardware
//
#include <hip/hip_runtime.h>
#include <hip/hip_bf16.h>

// ---------------------------------------------------------------------------
// Model constants
// ---------------------------------------------------------------------------
constexpr int SEQ    = 3072;
constexpr int HIDDEN = 1280;
constexpr int HEADS  = 16;
constexpr int HD     = 80;     // head dim
constexpr int HDP    = 96;     // head dim padded to multiple of 32 for WMMA K
constexpr int INTER  = 4608;
constexpr int NQKV   = 3 * HIDDEN; // 3840
constexpr int DEPTH  = 4;

// ---------------------------------------------------------------------------
// WMMA fragment types (CDNA5 wave32, v_wmma_f32_16x16x32_bf16)
// ---------------------------------------------------------------------------
typedef __attribute__((ext_vector_type(16))) __bf16 v16bf;
typedef __attribute__((ext_vector_type(8)))  __bf16 v8bf;
typedef __attribute__((ext_vector_type(8)))  float  v8f;
typedef __attribute__((ext_vector_type(4)))  unsigned int u32x4;
typedef __attribute__((ext_vector_type(8)))  int          i32x8;
typedef __attribute__((ext_vector_type(4)))  int          i32x4;

union ABFrag { v16bf v; v8bf h[2]; };

__device__ inline v8f wmma_bf16(v16bf a, v16bf b, v8f c) {
  // 8 args: (neg_a, A, neg_b, B, c_mod, C, reuse_a, reuse_b)
  return __builtin_amdgcn_wmma_f32_16x16x32_bf16(
      false, a, false, b, (short)0, c, false, false);
}

// ---------------------------------------------------------------------------
// Tensor Data Mover: stage a 2-D tile (tile_rows x tile_k bf16 elements,
// global row stride = row_stride elements) into LDS at byte offset lds_off.
// Descriptor bit layout per CDNA5 ISA.
// ---------------------------------------------------------------------------
#if defined(__has_builtin)
#if __has_builtin(__builtin_amdgcn_tensor_load_to_lds)
#define HAVE_TDM 1
#endif
#endif
#ifndef HAVE_TDM
#define HAVE_TDM 0
#endif

#if HAVE_TDM
__device__ inline void tdm_load_2d(unsigned lds_off, const __bf16* gptr,
                                   unsigned tile_k, unsigned tile_rows,
                                   unsigned row_stride) {
  unsigned long long ga = (unsigned long long)(size_t)gptr;
  u32x4 g0;
  g0[0] = 1u;                                   // count=1, user-mode D#
  g0[1] = lds_off;                              // lds_addr (bytes)
  g0[2] = (unsigned)(ga & 0xFFFFFFFFu);         // global_addr[31:0]
  g0[3] = (unsigned)((ga >> 32) & 0x1FFFFFFu)   // global_addr[56:32]
          | (2u << 30);                         // type = 2 ("image")
  i32x8 g1;
  unsigned td0 = row_stride;                    // tensor_dim0 (elems)
  unsigned td1 = 0x7FFFFFFFu;                   // tensor_dim1: large, no OOB
  unsigned long long st0 = (unsigned long long)row_stride; // dim0 stride
  g1[0] = (int)(1u << 16);                      // workgroup_mask=0, data_size=1 (2B)
  g1[1] = (int)((td0 & 0xFFFFu) << 16);         // tensor_dim0[15:0] @bit48
  g1[2] = (int)(((td0 >> 16) & 0xFFFFu) | ((td1 & 0xFFFFu) << 16));
  g1[3] = (int)(((td1 >> 16) & 0xFFFFu) | (tile_k << 16));   // tile_dim0
  g1[4] = (int)(tile_rows & 0xFFFFu);           // tile_dim1 (tile_dim2=0)
  g1[5] = (int)(st0 & 0xFFFFFFFFu);             // tensor_dim0_stride[31:0]
  g1[6] = (int)((st0 >> 32) & 0xFFFFu);         // stride[47:32], dim1 stride=0
  g1[7] = 0;
  i32x4 g2 = {0, 0, 0, 0};                      // 2-D tensor: dims 2/3 unused
  i32x4 g3 = {0, 0, 0, 0};
#if defined(__clang_major__) && __clang_major__ >= 23
  i32x8 g4 = {0, 0, 0, 0, 0, 0, 0, 0};
  __builtin_amdgcn_tensor_load_to_lds(g0, g1, g2, g3, g4, 0);
#else
  __builtin_amdgcn_tensor_load_to_lds(g0, g1, g2, g3, 0);
#endif
}
#endif // HAVE_TDM

// ---------------------------------------------------------------------------
// Utility kernels
// ---------------------------------------------------------------------------
__global__ void copyf_kernel(const float* __restrict__ src,
                             float* __restrict__ dst, int n) {
  int i = blockIdx.x * 256 + threadIdx.x;
  if (i < n) dst[i] = src[i];
}

// W is (K x N) row-major fp32; Wt is (N x K) row-major bf16.
__global__ void cast_transpose_kernel(const float* __restrict__ W,
                                      __bf16* __restrict__ Wt,
                                      int K, int N) {
  int idx = blockIdx.x * 256 + threadIdx.x;
  if (idx >= N * K) return;
  int n = idx / K;
  int k = idx - n * K;
  Wt[idx] = (__bf16)W[(size_t)k * N + n];
}

// LayerNorm over HIDDEN=1280 (= 5*256), one block per row, bf16 output.
__global__ __launch_bounds__(256)
void layernorm_bf16_kernel(const float* __restrict__ x,
                           const float* __restrict__ w,
                           const float* __restrict__ b,
                           __bf16* __restrict__ out) {
  __shared__ float red[256];
  const int row = blockIdx.x;
  const int tid = threadIdx.x;
  const float* xr = x + (size_t)row * HIDDEN;
  float v[5];
  float s = 0.f;
  for (int i = 0; i < 5; ++i) { v[i] = xr[tid + i * 256]; s += v[i]; }
  red[tid] = s; __syncthreads();
  for (int off = 128; off > 0; off >>= 1) {
    if (tid < off) red[tid] += red[tid + off];
    __syncthreads();
  }
  const float mu = red[0] * (1.f / HIDDEN);
  __syncthreads();
  s = 0.f;
  for (int i = 0; i < 5; ++i) { float d = v[i] - mu; s += d * d; }
  red[tid] = s; __syncthreads();
  for (int off = 128; off > 0; off >>= 1) {
    if (tid < off) red[tid] += red[tid + off];
    __syncthreads();
  }
  const float rstd = rsqrtf(red[0] * (1.f / HIDDEN) + 1e-6f);
  for (int i = 0; i < 5; ++i) {
    int c = tid + i * 256;
    out[(size_t)row * HIDDEN + c] = (__bf16)((v[i] - mu) * rstd * w[c] + b[c]);
  }
}

// Apply rotary to q,k from the fp32 qkv buffer; pack q,k head-major padded to
// 96 (zeros beyond 80) and v transposed (HEADS x 80 x SEQ), all bf16.
__global__ void rotary_pack_kernel(const float* __restrict__ qkv,
                                   const float* __restrict__ rot,
                                   __bf16* __restrict__ qb,
                                   __bf16* __restrict__ kb,
                                   __bf16* __restrict__ vt) {
  int idx = blockIdx.x * 256 + threadIdx.x; // SEQ*HEADS*40
  if (idx >= SEQ * HEADS * 40) return;
  int j = idx % 40;
  int t = idx / 40;
  int head = t % HEADS;
  int s = t / HEADS;
  const float* row = qkv + (size_t)s * NQKV;
  float ang = rot[s * 40 + j];
  float cs = cosf(ang), sn = sinf(ang);
  int base = head * HD + j;
  float q1 = row[base],              q2 = row[base + 40];
  float k1 = row[HIDDEN + base],     k2 = row[HIDDEN + base + 40];
  float v1 = row[2 * HIDDEN + base], v2 = row[2 * HIDDEN + base + 40];
  size_t qo = ((size_t)head * SEQ + s) * HDP;
  qb[qo + j]      = (__bf16)(q1 * cs - q2 * sn);
  qb[qo + j + 40] = (__bf16)(q2 * cs + q1 * sn);
  kb[qo + j]      = (__bf16)(k1 * cs - k2 * sn);
  kb[qo + j + 40] = (__bf16)(k2 * cs + k1 * sn);
  vt[((size_t)head * HD + j) * SEQ + s]      = (__bf16)v1;
  vt[((size_t)head * HD + j + 40) * SEQ + s] = (__bf16)v2;
  if (j < 8) { // zero the K padding 80..95
    qb[qo + 80 + j] = (__bf16)0.f; qb[qo + 88 + j] = (__bf16)0.f;
    kb[qo + 80 + j] = (__bf16)0.f; kb[qo + 88 + j] = (__bf16)0.f;
  }
}

// ---------------------------------------------------------------------------
// bf16 WMMA GEMM with TDM-staged, double-buffered B tiles in LDS.
//   C(MxN,f32) = A(MxK,bf16) * Bt(NxK,bf16)^T + bias
// Block: 256 threads = 8 waves; wave tile 32x32; block tile 128x64.
// K is consumed in steps of 64: B tile (64 cols x 64 K-elems, 8KB) is DMA'd
// into LDS by the Tensor Data Mover (wave 0 issues; TENSORcnt sync), while
// A fragments stream from global (L1/L2 resident).
// mode 0: outF = C+bias; mode 1: outF += C+bias;
// mode 2: outB = bf16(silu(aux) * (C+bias))   (fused SwiGLU, aux = gate f32)
// ---------------------------------------------------------------------------
__global__ __launch_bounds__(256)
void gemm_bf16_kernel(const __bf16* __restrict__ A,
                      const __bf16* __restrict__ Bt,
                      const float* __restrict__ bias,
                      int M, int N, int K, int mode,
                      float* __restrict__ outF,
                      __bf16* __restrict__ outB,
                      const float* __restrict__ aux) {
  __shared__ __bf16 bsm[2][64 * 64]; // double-buffered B tile, 2 x 8KB
  const int tid  = threadIdx.x;
  const int lane = tid & 31, wave = tid >> 5;
  const int lo = lane & 15, hi = lane >> 4;
  const int wm = wave >> 1, wn = wave & 1;
  const int m0 = blockIdx.x * 128 + wm * 32;
  const int n0 = blockIdx.y * 64;
  const int cn0 = wn * 32 + lo;       // tile-local column of b0 fragment
  const int cn1 = wn * 32 + 16 + lo;  // tile-local column of b1 fragment

  v8f c[2][2] = {};
  const __bf16* a0p   = A + (size_t)(m0 + lo)      * K + 16 * hi;
  const __bf16* a1p   = A + (size_t)(m0 + 16 + lo) * K + 16 * hi;
  const __bf16* btile = Bt + (size_t)n0 * K;
  const int NK = K / 64;

#if HAVE_TDM
  if (wave == 0) {
    tdm_load_2d((unsigned)(size_t)&bsm[0][0], btile, 64, 64, (unsigned)K);
  }
#endif
  for (int kt = 0; kt < NK; ++kt) {
    const int kbase = kt * 64;
#if HAVE_TDM
    if (wave == 0) {
      if (kt + 1 < NK) {
        tdm_load_2d((unsigned)(size_t)&bsm[(kt + 1) & 1][0],
                    btile + (size_t)(kt + 1) * 64, 64, 64, (unsigned)K);
        __builtin_amdgcn_s_wait_tensorcnt((short)1); // tile kt has landed
      } else {
        __builtin_amdgcn_s_wait_tensorcnt((short)0);
      }
    }
#else
    { // fallback: cooperative global->LDS staging (16 elems per thread)
      const int e   = tid * 16;
      const int row = e >> 6, col = e & 63;
      const __bf16* src = btile + (size_t)row * K + kbase + col;
      __bf16* dst = &bsm[kt & 1][row * 64 + col];
      *(v8bf*)dst       = *(const v8bf*)src;
      *(v8bf*)(dst + 8) = *(const v8bf*)(src + 8);
    }
#endif
    __syncthreads();
    const __bf16* bs = &bsm[kt & 1][0];
    for (int kk = 0; kk < 64; kk += 32) {
      ABFrag a0, a1;
      a0.h[0] = *(const v8bf*)(a0p + kbase + kk);
      a0.h[1] = *(const v8bf*)(a0p + kbase + kk + 8);
      a1.h[0] = *(const v8bf*)(a1p + kbase + kk);
      a1.h[1] = *(const v8bf*)(a1p + kbase + kk + 8);
      v16bf b0 = *(const v16bf*)(bs + cn0 * 64 + kk + 16 * hi);
      v16bf b1 = *(const v16bf*)(bs + cn1 * 64 + kk + 16 * hi);
      c[0][0] = wmma_bf16(a0.v, b0, c[0][0]);
      c[0][1] = wmma_bf16(a0.v, b1, c[0][1]);
      c[1][0] = wmma_bf16(a1.v, b0, c[1][0]);
      c[1][1] = wmma_bf16(a1.v, b1, c[1][1]);
    }
    __syncthreads(); // B tile fully consumed before TDM overwrites it
  }

  for (int im = 0; im < 2; ++im)
    for (int jn = 0; jn < 2; ++jn) {
      const int nn = n0 + wn * 32 + jn * 16 + lo;
      const float bv = bias ? bias[nn] : 0.f;
      for (int j = 0; j < 8; ++j) {
        const int mm = m0 + im * 16 + j + 8 * hi;
        const size_t idx = (size_t)mm * N + nn;
        float v = c[im][jn][j] + bv;
        if (mode == 0) {
          outF[idx] = v;
        } else if (mode == 1) {
          outF[idx] += v;
        } else {
          float g = aux[idx];
          float sg = g / (1.f + __expf(-g)); // silu(gate)
          outB[idx] = (__bf16)(sg * v);
        }
      }
    }
}

// ---------------------------------------------------------------------------
// Flash attention with sinks. Grid (48 qtiles, 16 heads), block 128 (4 waves).
// Online softmax state initialized with m = s_aux[head], l = 1 (sink term).
// full=0: single key tile (= query window); full=1: stream all 48 key tiles.
// ---------------------------------------------------------------------------
__global__ __launch_bounds__(128)
void attn_flash_kernel(const __bf16* __restrict__ qb,
                       const __bf16* __restrict__ kb,
                       const __bf16* __restrict__ vt,
                       const float* __restrict__ s_aux,
                       __bf16* __restrict__ obuf, int full) {
  __shared__ float  sc_s[64 * 64];
  __shared__ __bf16 p_s[64 * 64];
  __shared__ float  m_s[64], l_s[64], al_s[64];

  const int qt = blockIdx.x, head = blockIdx.y;
  const int tid = threadIdx.x;
  const int lane = tid & 31, wave = tid >> 5;
  const int lo = lane & 15, hi = lane >> 4;
  const float scale = 0.11180339887498949f; // 1/sqrt(80)

  const int qrow = qt * 64 + wave * 16 + lo;
  const __bf16* qp = qb + ((size_t)head * SEQ + qrow) * HDP + 16 * hi;
  ABFrag qf[3];
  for (int kk = 0; kk < 3; ++kk) {
    qf[kk].h[0] = *(const v8bf*)(qp + kk * 32);
    qf[kk].h[1] = *(const v8bf*)(qp + kk * 32 + 8);
  }

  if (tid < 64) { m_s[tid] = s_aux[head]; l_s[tid] = 1.0f; }
  __syncthreads();

  v8f of[5] = {};

  const int kt0 = full ? 0 : qt;
  const int kt1 = full ? SEQ / 64 : qt + 1;
  for (int kt = kt0; kt < kt1; ++kt) {
    const int sk0 = kt * 64;
    for (int ct = 0; ct < 4; ++ct) {
      const int kcol = sk0 + ct * 16 + lo;
      const __bf16* kp = kb + ((size_t)head * SEQ + kcol) * HDP + 16 * hi;
      v8f sc = {};
      for (int kk = 0; kk < 3; ++kk) {
        v16bf bf = *(const v16bf*)(kp + kk * 32);
        sc = wmma_bf16(qf[kk].v, bf, sc);
      }
      for (int j = 0; j < 8; ++j)
        sc_s[(wave * 16 + j + 8 * hi) * 64 + ct * 16 + lo] = sc[j] * scale;
    }
    __syncthreads();
    if (tid < 64) {
      const int r = tid;
      float mold = m_s[r], mx = mold;
      for (int c = 0; c < 64; ++c) mx = fmaxf(mx, sc_s[r * 64 + c]);
      float alpha = __expf(mold - mx);
      float sum = 0.f;
      for (int c = 0; c < 64; ++c) {
        float p = __expf(sc_s[r * 64 + c] - mx);
        p_s[r * 64 + c] = (__bf16)p;
        sum += p;
      }
      l_s[r] = l_s[r] * alpha + sum;
      m_s[r] = mx;
      al_s[r] = alpha;
    }
    __syncthreads();
    float al[8];
    for (int j = 0; j < 8; ++j) al[j] = al_s[wave * 16 + j + 8 * hi];
    for (int t = 0; t < 5; ++t) {
      v8f acc = of[t];
      for (int j = 0; j < 8; ++j) acc[j] *= al[j];
      for (int kk = 0; kk < 2; ++kk) {
        ABFrag pf;
        const __bf16* pp = &p_s[(wave * 16 + lo) * 64 + kk * 32 + 16 * hi];
        pf.h[0] = *(const v8bf*)pp;
        pf.h[1] = *(const v8bf*)(pp + 8);
        const __bf16* vp = vt + ((size_t)head * HD + t * 16 + lo) * SEQ
                              + sk0 + kk * 32 + 16 * hi;
        v16bf vf = *(const v16bf*)vp;
        acc = wmma_bf16(pf.v, vf, acc);
      }
      of[t] = acc;
    }
    __syncthreads();
  }

  for (int t = 0; t < 5; ++t)
    for (int j = 0; j < 8; ++j) {
      const int m = wave * 16 + j + 8 * hi;
      const int s = qt * 64 + m;
      const float denom = l_s[m];
      obuf[(size_t)s * HIDDEN + head * HD + t * 16 + lo] =
          (__bf16)(of[t][j] / denom);
    }
}

// ---------------------------------------------------------------------------
// Host-side launch
// ---------------------------------------------------------------------------
extern "C" void kernel_launch(void* const* d_in, const int* in_sizes, int n_in,
                              void* d_out, int out_size, void* d_ws,
                              size_t ws_size, hipStream_t stream) {
  (void)in_sizes; (void)n_in; (void)out_size; (void)ws_size;
  const float* x      = (const float*)d_in[0];
  const float* rot    = (const float*)d_in[1];
  const float* s_aux  = (const float*)d_in[2];
  const float* ln1_w  = (const float*)d_in[3];
  const float* ln1_b  = (const float*)d_in[4];
  const float* qkv_w  = (const float*)d_in[5];
  const float* qkv_b  = (const float*)d_in[6];
  const float* proj_w = (const float*)d_in[7];
  const float* proj_b = (const float*)d_in[8];
  const float* ln2_w  = (const float*)d_in[9];
  const float* ln2_b  = (const float*)d_in[10];
  const float* gate_w = (const float*)d_in[11];
  const float* gate_b = (const float*)d_in[12];
  const float* up_w   = (const float*)d_in[13];
  const float* up_b   = (const float*)d_in[14];
  const float* down_w = (const float*)d_in[15];
  const float* down_b = (const float*)d_in[16];

  char* ws = (char*)d_ws;
  size_t off = 0;
  auto alloc = [&](size_t bytes) -> void* {
    void* p = ws + off;
    off = (off + bytes + 255) & ~(size_t)255;
    return p;
  };
  __bf16* qkvw_t  = (__bf16*)alloc((size_t)NQKV * HIDDEN * 2);
  __bf16* projw_t = (__bf16*)alloc((size_t)HIDDEN * HIDDEN * 2);
  __bf16* gatew_t = (__bf16*)alloc((size_t)INTER * HIDDEN * 2);
  __bf16* upw_t   = (__bf16*)alloc((size_t)INTER * HIDDEN * 2);
  __bf16* downw_t = (__bf16*)alloc((size_t)HIDDEN * INTER * 2);
  __bf16* hnb    = (__bf16*)alloc((size_t)SEQ * HIDDEN * 2);
  float*  qkvbuf = (float*) alloc((size_t)SEQ * NQKV * 4);
  __bf16* qb     = (__bf16*)alloc((size_t)HEADS * SEQ * HDP * 2);
  __bf16* kbuf   = (__bf16*)alloc((size_t)HEADS * SEQ * HDP * 2);
  __bf16* vt     = (__bf16*)alloc((size_t)HEADS * HD * SEQ * 2);
  __bf16* obuf   = (__bf16*)alloc((size_t)SEQ * HIDDEN * 2);
  float*  gbuf   = (float*) alloc((size_t)SEQ * INTER * 4);
  __bf16* actb   = (__bf16*)alloc((size_t)SEQ * INTER * 2);

  float* h = (float*)d_out; // residual stream lives in d_out

  const int nEl = SEQ * HIDDEN;
  copyf_kernel<<<(nEl + 255) / 256, 256, 0, stream>>>(x, h, nEl);

  auto tgrid = [](int elems) { return dim3((elems + 255) / 256); };

  for (int l = 0; l < DEPTH; ++l) {
    cast_transpose_kernel<<<tgrid(HIDDEN * NQKV), 256, 0, stream>>>(
        qkv_w + (size_t)l * HIDDEN * NQKV, qkvw_t, HIDDEN, NQKV);
    cast_transpose_kernel<<<tgrid(HIDDEN * HIDDEN), 256, 0, stream>>>(
        proj_w + (size_t)l * HIDDEN * HIDDEN, projw_t, HIDDEN, HIDDEN);
    cast_transpose_kernel<<<tgrid(HIDDEN * INTER), 256, 0, stream>>>(
        gate_w + (size_t)l * HIDDEN * INTER, gatew_t, HIDDEN, INTER);
    cast_transpose_kernel<<<tgrid(HIDDEN * INTER), 256, 0, stream>>>(
        up_w + (size_t)l * HIDDEN * INTER, upw_t, HIDDEN, INTER);
    cast_transpose_kernel<<<tgrid(INTER * HIDDEN), 256, 0, stream>>>(
        down_w + (size_t)l * INTER * HIDDEN, downw_t, INTER, HIDDEN);

    layernorm_bf16_kernel<<<SEQ, 256, 0, stream>>>(
        h, ln1_w + l * HIDDEN, ln1_b + l * HIDDEN, hnb);
    {
      dim3 g(SEQ / 128, NQKV / 64);
      gemm_bf16_kernel<<<g, 256, 0, stream>>>(
          hnb, qkvw_t, qkv_b + (size_t)l * NQKV, SEQ, NQKV, HIDDEN,
          /*mode=*/0, qkvbuf, nullptr, nullptr);
    }
    rotary_pack_kernel<<<tgrid(SEQ * HEADS * 40), 256, 0, stream>>>(
        qkvbuf, rot, qb, kbuf, vt);
    {
      dim3 g(SEQ / 64, HEADS);
      attn_flash_kernel<<<g, 128, 0, stream>>>(
          qb, kbuf, vt, s_aux, obuf, (l == 3) ? 1 : 0);
    }
    {
      dim3 g(SEQ / 128, HIDDEN / 64);
      gemm_bf16_kernel<<<g, 256, 0, stream>>>(
          obuf, projw_t, proj_b + (size_t)l * HIDDEN, SEQ, HIDDEN, HIDDEN,
          /*mode=*/1, h, nullptr, nullptr);
    }

    layernorm_bf16_kernel<<<SEQ, 256, 0, stream>>>(
        h, ln2_w + l * HIDDEN, ln2_b + l * HIDDEN, hnb);
    {
      dim3 g(SEQ / 128, INTER / 64);
      gemm_bf16_kernel<<<g, 256, 0, stream>>>(
          hnb, gatew_t, gate_b + (size_t)l * INTER, SEQ, INTER, HIDDEN,
          /*mode=*/0, gbuf, nullptr, nullptr);
      gemm_bf16_kernel<<<g, 256, 0, stream>>>(
          hnb, upw_t, up_b + (size_t)l * INTER, SEQ, INTER, HIDDEN,
          /*mode=*/2, nullptr, actb, gbuf);
    }
    {
      dim3 g(SEQ / 128, HIDDEN / 64);
      gemm_bf16_kernel<<<g, 256, 0, stream>>>(
          actb, downw_t, down_b + (size_t)l * HIDDEN, SEQ, HIDDEN, INTER,
          /*mode=*/1, h, nullptr, nullptr);
    }
  }
}
